// GCNNet_2370821947640
// MI455X (gfx1250) — matrix operand order repaired and checked by reading, not compiled, not run on hardware
//
#include <hip/hip_runtime.h>
#include <hip/hip_bf16.h>

#define N_NODES  100000
#define N_EDGES  1600000
#define N_GRAPHS 1024
#define NODE_IN  163
#define KPAD0    192
#define HIDDEN   64
#define SLOPE    0.01f

typedef __attribute__((ext_vector_type(16))) __bf16 v16bf;
typedef __attribute__((ext_vector_type(8)))  float  v8f;

static __device__ __forceinline__ float lrelu(float v) {
    return v > 0.0f ? v : v * SLOPE;
}

// round-to-nearest-even f32 -> bf16 (bit pattern)
static __device__ __forceinline__ unsigned short f32_to_bf16(float f) {
    unsigned int u = __float_as_uint(f);
    u += 0x7FFFu + ((u >> 16) & 1u);
    return (unsigned short)(u >> 16);
}

// ---------------------------------------------------------------- utilities
__global__ void zero_f32(float* p, int n) {
    int i = blockIdx.x * blockDim.x + threadIdx.x;
    if (i < n) p[i] = 0.0f;
}

__global__ void deg_kernel(const int* __restrict__ dst, float* __restrict__ deg) {
    int e = blockIdx.x * blockDim.x + threadIdx.x;
    if (e < N_EDGES) atomicAdd(&deg[dst[e]], 1.0f);
}

__global__ void dis_kernel(float* __restrict__ deg_to_dis) {
    int i = blockIdx.x * blockDim.x + threadIdx.x;
    if (i < N_NODES) deg_to_dis[i] = rsqrtf(deg_to_dis[i] + 1.0f);
}

__global__ void enorm_kernel(const int* __restrict__ src, const int* __restrict__ dst,
                             const float* __restrict__ dis, float* __restrict__ enorm) {
    int e = blockIdx.x * blockDim.x + threadIdx.x;
    if (e < N_EDGES) enorm[e] = dis[src[e]] * dis[dst[e]];
}

// f32 [N x K] -> zero-padded bf16 [N x Kp]
__global__ void cvt_pad_bf16(const float* __restrict__ A, unsigned short* __restrict__ Abf,
                             int K, int Kp, int total) {
    int idx = blockIdx.x * blockDim.x + threadIdx.x;
    if (idx >= total) return;
    int i = idx / Kp;
    int k = idx - i * Kp;
    Abf[idx] = (k < K) ? f32_to_bf16(A[(size_t)i * K + k]) : (unsigned short)0;
}

// -------------------------------------------------- weight pre-swizzle (B frag)
// frag[((kt*4 + nt)*32 + lane)*16 + i]: lane&15 = column N in tile, lane>>4 = K half,
// element i covers K = kt*32 + half*16 + i.
__global__ void wswizzle_kernel(const float* __restrict__ W, unsigned short* __restrict__ frag,
                                int K) {
    int b    = blockIdx.x;            // kt*4 + nt
    int kt   = b >> 2;
    int nt   = b & 3;
    int lane = threadIdx.x;           // 0..31
    int col  = nt * 16 + (lane & 15);
    int kb   = kt * 32 + (lane >> 4) * 16;
    unsigned short* out = frag + ((size_t)b * 32 + lane) * 16;
#pragma unroll
    for (int i = 0; i < 16; ++i) {
        int k = kb + i;
        float f = (k < K) ? W[(size_t)k * HIDDEN + col] : 0.0f;
        out[i] = f32_to_bf16(f);
    }
}

// -------------------------------------------------------------- WMMA GEMM
// C[N_NODES x 64] = Abf[N_NODES x Kp] @ W. One wave computes a 16x64 strip:
// a single A fragment feeds 4 chained v_wmma_f32_16x16x32_bf16 per k-step.
__global__ void __launch_bounds__(256)
gemm_wmma(const unsigned short* __restrict__ Abf, const unsigned short* __restrict__ Bfrag,
          float* __restrict__ C, int Kp, int ktiles) {
    const int lane  = threadIdx.x & 31;
    const int mtile = blockIdx.x * 8 + (threadIdx.x >> 5);
    if (mtile >= N_NODES / 16) return;             // wave-uniform: EXEC all-ones inside

    const int m0   = mtile * 16;
    const int half = lane >> 4;
    const int row  = m0 + (lane & 15);
    const unsigned short* __restrict__ arow = Abf + (size_t)row * Kp;

    v8f acc0 = {}, acc1 = {}, acc2 = {}, acc3 = {};
    for (int kt = 0; kt < ktiles; ++kt) {
        const int k0 = kt * 32;
        // A fragment: element i covers K = k0 + half*16 + i (matches B frag)
        union { v16bf v; uint4 q[2]; } af;
        af.q[0] = *(const uint4*)(arow + k0 + half * 16);
        af.q[1] = *(const uint4*)(arow + k0 + half * 16 + 8);

        const unsigned short* __restrict__ bb =
            Bfrag + ((size_t)kt * 4 * 32 + lane) * 16;     // nt stride = 512 elements
        union { v16bf v; uint4 q[2]; } bf0, bf1, bf2, bf3;
        bf0.q[0] = *(const uint4*)(bb +    0); bf0.q[1] = *(const uint4*)(bb +    8);
        bf1.q[0] = *(const uint4*)(bb +  512); bf1.q[1] = *(const uint4*)(bb +  520);
        bf2.q[0] = *(const uint4*)(bb + 1024); bf2.q[1] = *(const uint4*)(bb + 1032);
        bf3.q[0] = *(const uint4*)(bb + 1536); bf3.q[1] = *(const uint4*)(bb + 1544);

        acc0 = __builtin_amdgcn_wmma_f32_16x16x32_bf16(false, af.v, false, bf0.v, (short)0, acc0, false, false);
        acc1 = __builtin_amdgcn_wmma_f32_16x16x32_bf16(false, af.v, false, bf1.v, (short)0, acc1, false, false);
        acc2 = __builtin_amdgcn_wmma_f32_16x16x32_bf16(false, af.v, false, bf2.v, (short)0, acc2, false, false);
        acc3 = __builtin_amdgcn_wmma_f32_16x16x32_bf16(false, af.v, false, bf3.v, (short)0, acc3, false, false);
    }
    // store: VGPR r -> row m0+half*8+r, col = nt*16 + (lane&15)
    float* __restrict__ crow = C + (size_t)(m0 + half * 8) * HIDDEN + (lane & 15);
#pragma unroll
    for (int r = 0; r < 8; ++r) {
        crow[(size_t)r * HIDDEN +  0] = acc0[r];
        crow[(size_t)r * HIDDEN + 16] = acc1[r];
        crow[(size_t)r * HIDDEN + 32] = acc2[r];
        crow[(size_t)r * HIDDEN + 48] = acc3[r];
    }
}

// ------------------------------------------------------------ edge scatter
__global__ void scatter_kernel(const int* __restrict__ src, const int* __restrict__ dst,
                               const float* __restrict__ enorm,
                               const float* __restrict__ t, float* __restrict__ agg) {
    long long idx = (long long)blockIdx.x * blockDim.x + threadIdx.x;
    if (idx >= (long long)N_EDGES * HIDDEN) return;
    int e = (int)(idx >> 6);
    int j = (int)(idx & 63);
    int s = src[e], d = dst[e];
    atomicAdd(&agg[(size_t)d * HIDDEN + j], t[(size_t)s * HIDDEN + j] * enorm[e]);
}

// self-loop + bias + leaky relu in place on agg; also emit bf16 copy for next GEMM
__global__ void finalize_kernel(float* __restrict__ agg, const float* __restrict__ t,
                                const float* __restrict__ dis, const float* __restrict__ bias,
                                unsigned short* __restrict__ hbf) {
    int idx = blockIdx.x * blockDim.x + threadIdx.x;
    if (idx >= N_NODES * HIDDEN) return;
    int i = idx >> 6;
    int j = idx & 63;
    float sn = dis[i] * dis[i];
    float v = lrelu(agg[idx] + t[idx] * sn + bias[j]);
    agg[idx] = v;
    hbf[idx] = f32_to_bf16(v);
}

// ----------------------------------------------------------------- pooling
__global__ void count_kernel(const int* __restrict__ batch, float* __restrict__ counts) {
    int i = blockIdx.x * blockDim.x + threadIdx.x;
    if (i < N_NODES) atomicAdd(&counts[batch[i]], 1.0f);
}

__global__ void poolsum_kernel(const int* __restrict__ batch, const float* __restrict__ h,
                               float* __restrict__ pooled) {
    int idx = blockIdx.x * blockDim.x + threadIdx.x;
    if (idx >= N_NODES * HIDDEN) return;
    int i = idx >> 6;
    int j = idx & 63;
    atomicAdd(&pooled[(size_t)batch[i] * HIDDEN + j], h[idx]);
}

// ------------------------------------------------------------- graph MLP
__global__ void __launch_bounds__(64)
mlp_kernel(const float* __restrict__ pooled, const float* __restrict__ counts,
           const float* __restrict__ fc1W, const float* __restrict__ fc1b,
           const float* __restrict__ fc2W, const float* __restrict__ fc2b,
           float* __restrict__ out) {
    __shared__ float p[HIDDEN];
    __shared__ float red[HIDDEN];
    int g = blockIdx.x, j = threadIdx.x;
    float cnt = fmaxf(counts[g], 1.0f);
    p[j] = pooled[(size_t)g * HIDDEN + j] / cnt;
    __syncthreads();
    float acc = fc1b[j];
#pragma unroll 8
    for (int k = 0; k < HIDDEN; ++k)
        acc += p[k] * fc1W[(size_t)k * HIDDEN + j];
    red[j] = lrelu(acc) * fc2W[j];
    __syncthreads();
    for (int s = 32; s > 0; s >>= 1) {
        if (j < s) red[j] += red[j + s];
        __syncthreads();
    }
    if (j == 0) out[g] = red[0] + fc2b[0];
}

// ================================================================= launch
static inline char* carve(char*& p, size_t bytes) {
    char* r = p;
    p += (bytes + 255) & ~(size_t)255;
    return r;
}

extern "C" void kernel_launch(void* const* d_in, const int* in_sizes, int n_in,
                              void* d_out, int out_size, void* d_ws, size_t ws_size,
                              hipStream_t stream) {
    const float* x     = (const float*)d_in[0];
    const int*   eidx  = (const int*)d_in[1];
    const int*   batch = (const int*)d_in[2];
    const float* W0    = (const float*)d_in[3];
    const float* b0    = (const float*)d_in[4];
    const float* W1    = (const float*)d_in[5];
    const float* b1    = (const float*)d_in[6];
    const float* W2    = (const float*)d_in[7];
    const float* b2    = (const float*)d_in[8];
    const float* fc1W  = (const float*)d_in[9];
    const float* fc1b  = (const float*)d_in[10];
    const float* fc2W  = (const float*)d_in[11];
    const float* fc2b  = (const float*)d_in[12];
    const int* src = eidx;
    const int* dst = eidx + N_EDGES;
    float* out = (float*)d_out;

    char* ws = (char*)d_ws;
    float* dis   = (float*)carve(ws, sizeof(float) * N_NODES);            // deg -> dis in place
    float* enorm = (float*)carve(ws, sizeof(float) * N_EDGES);
    float* T     = (float*)carve(ws, sizeof(float) * (size_t)N_NODES * HIDDEN);
    float* AGG   = (float*)carve(ws, sizeof(float) * (size_t)N_NODES * HIDDEN);
    unsigned short* Abf0 = (unsigned short*)carve(ws, sizeof(unsigned short) * (size_t)N_NODES * KPAD0);
    unsigned short* AbfH = (unsigned short*)carve(ws, sizeof(unsigned short) * (size_t)N_NODES * HIDDEN);
    unsigned short* W0f  = (unsigned short*)carve(ws, sizeof(unsigned short) * 6 * 4 * 32 * 16);
    unsigned short* W1f  = (unsigned short*)carve(ws, sizeof(unsigned short) * 2 * 4 * 32 * 16);
    unsigned short* W2f  = (unsigned short*)carve(ws, sizeof(unsigned short) * 2 * 4 * 32 * 16);
    float* pooled = (float*)carve(ws, sizeof(float) * N_GRAPHS * HIDDEN);
    float* counts = (float*)carve(ws, sizeof(float) * N_GRAPHS);

    const int NH = N_NODES * HIDDEN;                  // 6.4M
    const long long EH = (long long)N_EDGES * HIDDEN; // 102.4M
    const int gemm_blocks = (N_NODES / 16 + 7) / 8;   // 782

    // ---- normalization terms ----
    zero_f32<<<(N_NODES + 255) / 256, 256, 0, stream>>>(dis, N_NODES);
    deg_kernel<<<(N_EDGES + 255) / 256, 256, 0, stream>>>(dst, dis);
    dis_kernel<<<(N_NODES + 255) / 256, 256, 0, stream>>>(dis);
    enorm_kernel<<<(N_EDGES + 255) / 256, 256, 0, stream>>>(src, dst, dis, enorm);

    // ---- operand preparation ----
    wswizzle_kernel<<<6 * 4, 32, 0, stream>>>(W0, W0f, NODE_IN);
    wswizzle_kernel<<<2 * 4, 32, 0, stream>>>(W1, W1f, HIDDEN);
    wswizzle_kernel<<<2 * 4, 32, 0, stream>>>(W2, W2f, HIDDEN);
    {
        int total = N_NODES * KPAD0;                  // 19.2M
        cvt_pad_bf16<<<(total + 255) / 256, 256, 0, stream>>>(x, Abf0, NODE_IN, KPAD0, total);
    }

    // ---- layer 0 ----
    gemm_wmma<<<gemm_blocks, 256, 0, stream>>>(Abf0, W0f, T, KPAD0, 6);
    zero_f32<<<(NH + 255) / 256, 256, 0, stream>>>(AGG, NH);
    scatter_kernel<<<(int)((EH + 255) / 256), 256, 0, stream>>>(src, dst, enorm, T, AGG);
    finalize_kernel<<<(NH + 255) / 256, 256, 0, stream>>>(AGG, T, dis, b0, AbfH);

    // ---- layer 1 ----
    gemm_wmma<<<gemm_blocks, 256, 0, stream>>>(AbfH, W1f, T, HIDDEN, 2);
    zero_f32<<<(NH + 255) / 256, 256, 0, stream>>>(AGG, NH);
    scatter_kernel<<<(int)((EH + 255) / 256), 256, 0, stream>>>(src, dst, enorm, T, AGG);
    finalize_kernel<<<(NH + 255) / 256, 256, 0, stream>>>(AGG, T, dis, b1, AbfH);

    // ---- layer 2 ----
    gemm_wmma<<<gemm_blocks, 256, 0, stream>>>(AbfH, W2f, T, HIDDEN, 2);
    zero_f32<<<(NH + 255) / 256, 256, 0, stream>>>(AGG, NH);
    scatter_kernel<<<(int)((EH + 255) / 256), 256, 0, stream>>>(src, dst, enorm, T, AGG);
    finalize_kernel<<<(NH + 255) / 256, 256, 0, stream>>>(AGG, T, dis, b2, AbfH);

    // ---- pooling + MLP ----
    zero_f32<<<(N_GRAPHS * HIDDEN + 255) / 256, 256, 0, stream>>>(pooled, N_GRAPHS * HIDDEN);
    zero_f32<<<(N_GRAPHS + 255) / 256, 256, 0, stream>>>(counts, N_GRAPHS);
    count_kernel<<<(N_NODES + 255) / 256, 256, 0, stream>>>(batch, counts);
    poolsum_kernel<<<(NH + 255) / 256, 256, 0, stream>>>(batch, AGG, pooled);
    mlp_kernel<<<N_GRAPHS, 64, 0, stream>>>(pooled, counts, fc1W, fc1b, fc2W, fc2b, out);
}
